// FourierGlobalAttention_13572096656028
// MI455X (gfx1250) — matrix operand order repaired and checked by reading, not compiled, not run on hardware
//
#include <hip/hip_runtime.h>
#include <hip/hip_bf16.h>
#include <math.h>

typedef __attribute__((ext_vector_type(16))) __bf16       v16bf;
typedef __attribute__((ext_vector_type(8)))  float        v8f;
typedef __attribute__((ext_vector_type(4)))  int          v4i;
typedef __attribute__((ext_vector_type(4)))  unsigned int v4u;
typedef __attribute__((ext_vector_type(8)))  int          v8i;

#define BATCH    8
#define CH       512
#define NHEADS   8
#define HEADD    64
#define NTOK     1024          // tokens per batch (32*32)
#define TTOT     8192          // BATCH * NTOK
#define QKV_N    1536
#define SCALE_F  0.125f        // 64^-0.5
#define PI_F     3.14159265358979323846f

#define GLOBAL_AS __attribute__((address_space(1)))
#define LDS_AS    __attribute__((address_space(3)))

static __device__ __forceinline__ __bf16 f2bf(float f) { return (__bf16)f; }

static __device__ __forceinline__ v8f wmma_bf16(v16bf a, v16bf b, v8f c) {
  // D = A(16x32 bf16) x B(32x16 bf16) + C(16x16 f32)
  return __builtin_amdgcn_wmma_f32_16x16x32_bf16(false, a, false, b, (short)0, c, false, false);
}

// ---- CDNA5 async global->LDS copy (ASYNCcnt-tracked) ----------------------
#if defined(__HIP_DEVICE_COMPILE__) && __has_builtin(__builtin_amdgcn_global_load_async_to_lds_b128)
#define HAVE_ASYNC_LDS 1
static __device__ __forceinline__ void async_copy_b128(const __bf16* g, __bf16* l) {
  __builtin_amdgcn_global_load_async_to_lds_b128(
      (GLOBAL_AS v4i*)(v4i*)const_cast<__bf16*>(g),
      (LDS_AS v4i*)(v4i*)l, 0, 0);
}
static __device__ __forceinline__ void wait_async0() {
#if __has_builtin(__builtin_amdgcn_s_wait_asynccnt)
  __builtin_amdgcn_s_wait_asynccnt(0);
#else
  asm volatile("s_wait_asynccnt 0" ::: "memory");
#endif
}
#else
#define HAVE_ASYNC_LDS 0
#endif

// ---- CDNA5 Tensor Data Mover: 1-D tile global->LDS (TENSORcnt-tracked) ----
#if defined(__HIP_DEVICE_COMPILE__) && __has_builtin(__builtin_amdgcn_tensor_load_to_lds)
#define HAVE_TDM 1
static __device__ __forceinline__ void wait_tensor0() {
#if __has_builtin(__builtin_amdgcn_s_wait_tensorcnt)
  __builtin_amdgcn_s_wait_tensorcnt(0);
#else
  asm volatile("s_wait_tensorcnt 0" ::: "memory");
#endif
}
// nelem 2-byte elements, contiguous; builds D# groups per ISA sect. 8.3/8.4.
static __device__ __forceinline__ void tdm_load_1d(const void* gsrc, void* lds,
                                                   unsigned int nelem) {
  unsigned long long ga = (unsigned long long)(size_t)gsrc;
  unsigned int la = (unsigned int)(size_t)(LDS_AS char*)lds;   // raw LDS byte addr
  v4u g0;
  g0[0] = 1u;                                                  // count=1 (valid, user)
  g0[1] = la;                                                  // lds_addr
  g0[2] = (unsigned int)ga;                                    // global_addr[31:0]
  g0[3] = ((unsigned int)(ga >> 32) & 0x01FFFFFFu) | 0x80000000u; // addr[56:32] | type=2
  v8i g1;
  g1[0] = 0x00010000;                  // wg_mask=0 (no cluster), data_size=2B
  g1[1] = (int)(nelem << 16);          // [47:32]=atomic_bar=0, [63:48]=tensor_dim0 lo
  g1[2] = (int)((nelem >> 16) | (1u << 16)); // [79:64]=tensor_dim0 hi, [95:80]=tensor_dim1=1
  g1[3] = (int)(nelem << 16);          // [111:96]=dim1 hi=0, [127:112]=tile_dim0
  g1[4] = 0;                           // tile_dim1=0 (unused), tile_dim2=0
  g1[5] = (int)nelem;                  // tensor_dim0_stride lo
  g1[6] = 0;
  g1[7] = 0;
  v4i z4 = {0, 0, 0, 0};
#if __has_include(<hip/amd_detail/amd_gfx1250_TDM.h>)
  v8i z8 = {0, 0, 0, 0, 0, 0, 0, 0};   // therock toolchain: 6-arg form
  __builtin_amdgcn_tensor_load_to_lds(g0, g1, z4, z4, z8, 0);
#else
  __builtin_amdgcn_tensor_load_to_lds(g0, g1, z4, z4, 0);      // ROCm 7.2: 5-arg form
#endif
}
#else
#define HAVE_TDM 0
#endif

// A-matrix fragment (16x32 bf16): per lane 8 contiguous bf16 at k=8*half and
// 8 at k=16+8*half, row = lane&15.
static __device__ __forceinline__ v16bf load_a_frag(const __bf16* tile, int row0,
                                                    int stride, int lane) {
  int m = lane & 15;
  int half = lane >> 4;
  const __bf16* p = tile + (size_t)(row0 + m) * stride;
  int b0 = half * 8;
  int b1 = 16 + half * 8;
  v16bf a{};
#pragma unroll
  for (int i = 0; i < 8; ++i) { a[i] = p[b0 + i]; a[8 + i] = p[b1 + i]; }
  return a;
}

// B-matrix fragment (32x16 bf16): col n = lane&15, per lane 8 at k=8*half and
// 8 at k=16+8*half (K-major tile).
static __device__ __forceinline__ v16bf load_b_frag(const __bf16* tileT, int k0,
                                                    int stride, int n0, int lane) {
  int half = lane >> 4;
  const __bf16* p = tileT + (size_t)k0 * stride + n0 + (lane & 15);
  v16bf b{};
#pragma unroll
  for (int i = 0; i < 8; ++i) {
    b[i] = p[(size_t)(half * 8 + i) * stride];
    b[8 + i] = p[(size_t)(16 + half * 8 + i) * stride];
  }
  return b;
}

// ---------------------------------------------------------------------------
// Kernel 0: T[h,w] = irfft2((1+1j)*ones(32,17), s=(32,32), norm='ortho')
// ---------------------------------------------------------------------------
__global__ void ttab_kernel(float* __restrict__ T) {
  int h = threadIdx.x >> 5, w = threadIdx.x & 31;
  float acc = 0.f;
  if (h == 0) {
    acc = 1.0f + cosf(PI_F * (float)w);
#pragma unroll
    for (int k = 1; k <= 15; ++k) {
      float th = (2.0f * PI_F) * (float)k * (float)w * (1.0f / 32.0f);
      acc += 2.0f * (cosf(th) - sinf(th));
    }
  }
  T[threadIdx.x] = acc;
}

// ---------------------------------------------------------------------------
// Kernel 1: QKV GEMM (8192x512)x(512x1536), 64x64 tile/block, K-step 64
// ---------------------------------------------------------------------------
__global__ __launch_bounds__(256)
void qkv_kernel(const float* __restrict__ x, const float* __restrict__ qkv_w,
                __bf16* __restrict__ qo, __bf16* __restrict__ ko, __bf16* __restrict__ vo) {
  __shared__ __bf16 At[64][64];   // M x K
  __shared__ __bf16 Bt[64][64];   // K x N (K-major)
  int tid = threadIdx.x, wave = tid >> 5, lane = tid & 31;
  int m0 = blockIdx.x * 64, n0 = blockIdx.y * 64;
  int r16 = (wave & 3) * 16;
  int cn  = (wave >> 2) * 32;
  v8f acc0 = {}, acc1 = {};

  for (int kk = 0; kk < CH; kk += 64) {
    { // stage A: x -> bf16 (16 per thread)
      int row = tid >> 2, c0 = (tid & 3) * 16;
      const float* src = x + (size_t)(m0 + row) * CH + kk + c0;
      __builtin_prefetch(src + 64, 0, 0);
#pragma unroll
      for (int i = 0; i < 16; ++i) At[row][c0 + i] = f2bf(src[i]);
    }
    { // stage B transposed: Bt[k][n] = qkv_w[n0+n][kk+k]
      int n = tid >> 2, c0 = (tid & 3) * 16;
      const float* src = qkv_w + (size_t)(n0 + n) * CH + kk + c0;
#pragma unroll
      for (int i = 0; i < 16; ++i) Bt[c0 + i][n] = f2bf(src[i]);
    }
    __syncthreads();
    v16bf a0 = load_a_frag(&At[0][0],  r16, 64, lane);
    v16bf a1 = load_a_frag(&At[0][32], r16, 64, lane);
    acc0 = wmma_bf16(a0, load_b_frag(&Bt[0][0], 0,  64, cn,      lane), acc0);
    acc0 = wmma_bf16(a1, load_b_frag(&Bt[0][0], 32, 64, cn,      lane), acc0);
    acc1 = wmma_bf16(a0, load_b_frag(&Bt[0][0], 0,  64, cn + 16, lane), acc1);
    acc1 = wmma_bf16(a1, load_b_frag(&Bt[0][0], 32, 64, cn + 16, lane), acc1);
    __syncthreads();
  }

  // scatter: base pointer is lane-uniform in r; r advances address by HEADD
  int nloc = lane & 15, half = lane >> 4;
  int b = m0 >> 10;                              // tile never straddles a batch
  int nb0 = (m0 & 1023) + r16 + 8 * half;
#pragma unroll
  for (int which = 0; which < 2; ++which) {
    int j = n0 + cn + which * 16 + nloc;         // 0..1535
    int s = j >> 9, rem = j & 511;
    int head = rem >> 6, d = rem & 63;
    __bf16* dst = ((s == 0) ? qo : (s == 1) ? ko : vo)
                + (((size_t)(b * NHEADS + head) * NTOK) + nb0) * HEADD + d;
    v8f acc = which ? acc1 : acc0;
#pragma unroll
    for (int r = 0; r < 8; ++r) dst[(size_t)r * HEADD] = f2bf(acc[r]);
  }
}

// ---------------------------------------------------------------------------
// Kernel 2: flash attention. 16 waves x 16 queries; 128-key chunks.
// V chunk staged by the Tensor Data Mover (one descriptor from wave 0);
// K chunk transposed into Kt[d][key] by all threads meanwhile.
// ---------------------------------------------------------------------------
#define KCHUNK 128

__global__ __launch_bounds__(512)
void attn_kernel(const __bf16* __restrict__ qb, const __bf16* __restrict__ kb,
                 const __bf16* __restrict__ vb, __bf16* __restrict__ ao) {
  __shared__ __bf16 Kt[HEADD][KCHUNK];     // 16 KB
  __shared__ __bf16 Vs[KCHUNK][HEADD];     // 16 KB
  __shared__ __bf16 Pbuf[16][16 * 32];     // 16 KB, per-wave P staging

  int bh = blockIdx.x >> 2;
  int qc = blockIdx.x & 3;
  int tid = threadIdx.x, wave = tid >> 5, lane = tid & 31;
  int qrow0 = qc * 256 + wave * 16;

  const __bf16* qbase = qb + (size_t)bh * NTOK * HEADD;
  const __bf16* kbase = kb + (size_t)bh * NTOK * HEADD;
  const __bf16* vbase = vb + (size_t)bh * NTOK * HEADD;

  v16bf qf0 = load_a_frag(qbase,      qrow0, HEADD, lane);
  v16bf qf1 = load_a_frag(qbase + 32, qrow0, HEADD, lane);

  float mrow[8], lrow[8];
  v8f o[4] = {{}, {}, {}, {}};
#pragma unroll
  for (int r = 0; r < 8; ++r) { mrow[r] = -1e30f; lrow[r] = 0.f; }

  __bf16* myP = &Pbuf[wave][0];
  int nloc = lane & 15, half = lane >> 4;

  for (int c = 0; c < NTOK / KCHUNK; ++c) {
    int idx = tid * 16;                    // 16 bf16 per thread (32 B)
    int kkey = idx >> 6, d0 = idx & 63;
    { // V: contiguous 16 KB tile -> TDM, else async, else scalar copy
#if HAVE_TDM
      if (wave == 0)
        tdm_load_1d(vbase + (size_t)c * KCHUNK * HEADD, &Vs[0][0], KCHUNK * HEADD);
#elif HAVE_ASYNC_LDS
      const __bf16* src = vbase + (size_t)(c * KCHUNK + kkey) * HEADD + d0;
      __bf16* dst = &Vs[kkey][d0];
      async_copy_b128(src, dst);
      async_copy_b128(src + 8, dst + 8);
#else
      const __bf16* src = vbase + (size_t)(c * KCHUNK + kkey) * HEADD + d0;
      __bf16* dst = &Vs[kkey][d0];
#pragma unroll
      for (int i = 0; i < 16; ++i) dst[i] = src[i];
#endif
    }
    { // K: b128 global read, transposed ds_store_b16 at immediate offsets
      const __bf16* src = kbase + (size_t)(c * KCHUNK + kkey) * HEADD + d0;
#pragma unroll
      for (int i = 0; i < 16; ++i) Kt[d0 + i][kkey] = src[i];
    }
#if HAVE_TDM
    wait_tensor0();                        // drain TENSORcnt before barrier
#elif HAVE_ASYNC_LDS
    wait_async0();                         // drain ASYNCcnt before barrier
#endif
    __syncthreads();

    for (int sb = 0; sb < KCHUNK / 32; ++sb) {
      int kcol = sb * 32;
      v8f s0 = {}, s1 = {};
      s0 = wmma_bf16(qf0, load_b_frag(&Kt[0][0], 0,  KCHUNK, kcol,      lane), s0);
      s1 = wmma_bf16(qf0, load_b_frag(&Kt[0][0], 0,  KCHUNK, kcol + 16, lane), s1);
      s0 = wmma_bf16(qf1, load_b_frag(&Kt[0][0], 32, KCHUNK, kcol,      lane), s0);
      s1 = wmma_bf16(qf1, load_b_frag(&Kt[0][0], 32, KCHUNK, kcol + 16, lane), s1);

#pragma unroll
      for (int r = 0; r < 8; ++r) {
        float a0 = s0[r] * SCALE_F, a1 = s1[r] * SCALE_F;
        float rv = fmaxf(a0, a1);
        rv = fmaxf(rv, __shfl_xor(rv, 1, 32));
        rv = fmaxf(rv, __shfl_xor(rv, 2, 32));
        rv = fmaxf(rv, __shfl_xor(rv, 4, 32));
        rv = fmaxf(rv, __shfl_xor(rv, 8, 32));
        float mnew = fmaxf(mrow[r], rv);
        float alpha = __expf(mrow[r] - mnew);
        mrow[r] = mnew;
        float p0 = __expf(a0 - mnew);
        float p1 = __expf(a1 - mnew);
        float rs = p0 + p1;
        rs += __shfl_xor(rs, 1, 32);
        rs += __shfl_xor(rs, 2, 32);
        rs += __shfl_xor(rs, 4, 32);
        rs += __shfl_xor(rs, 8, 32);
        lrow[r] = lrow[r] * alpha + rs;
        o[0][r] *= alpha; o[1][r] *= alpha; o[2][r] *= alpha; o[3][r] *= alpha;
        s0[r] = p0; s1[r] = p1;
      }

      // D-layout -> A-layout via wave-private LDS staging (in-order per wave)
#pragma unroll
      for (int r = 0; r < 8; ++r) {
        myP[(r + 8 * half) * 32 + nloc]      = f2bf(s0[r]);
        myP[(r + 8 * half) * 32 + 16 + nloc] = f2bf(s1[r]);
      }
      v16bf pa = load_a_frag(myP, 0, 32, lane);
#pragma unroll
      for (int nt = 0; nt < 4; ++nt)
        o[nt] = wmma_bf16(pa, load_b_frag(&Vs[0][0], kcol, HEADD, nt * 16, lane), o[nt]);
    }
    __syncthreads();
  }

  // normalize + store [t][c] with hoisted base (immediate offsets over r)
  int head = bh & 7, b = bh >> 3;
  float invl[8];
#pragma unroll
  for (int r = 0; r < 8; ++r) invl[r] = 1.f / lrow[r];
  __bf16* obase = ao + (size_t)(b * NTOK + qrow0 + 8 * half) * CH + head * HEADD + nloc;
#pragma unroll
  for (int nt = 0; nt < 4; ++nt)
#pragma unroll
    for (int r = 0; r < 8; ++r)
      obase[(size_t)r * CH + nt * 16] = f2bf(o[nt][r] * invl[r]);
}

// ---------------------------------------------------------------------------
// Kernel 3: proj GEMM + analytic Fourier + LayerNorm + mix.
// Full 64x512 bf16 A-tile resident in LDS (async-staged once), B staged per
// (nb, kk) with K-step 64. g/bb/T cached in LDS.
// ---------------------------------------------------------------------------
__global__ __launch_bounds__(256)
void final_kernel(const __bf16* __restrict__ ao, const float* __restrict__ x,
                  const float* __restrict__ proj_w, const float* __restrict__ proj_b,
                  const float* __restrict__ fconv_w, const float* __restrict__ fconv_b,
                  const float* __restrict__ freq_w, const float* __restrict__ ln_g,
                  const float* __restrict__ ln_b, const float* __restrict__ mix_w,
                  const float* __restrict__ Ttab, float* __restrict__ out) {
  __shared__ __bf16 Afull[64][512];   // 64 KB
  __shared__ __bf16 Bt[64][64];       // 8 KB (K-major)
  __shared__ float g_s[512], bb_s[512];
  __shared__ float T_s[64], mu_s[64], rs_s[64];
  __shared__ float red[256];

  int tid = threadIdx.x, wave = tid >> 5, lane = tid & 31;
  int m0 = blockIdx.x * 64;

  // ---- preamble: stage A (async), g/bb, per-token T
#pragma unroll
  for (int chk = 0; chk < 8; ++chk) {
    int idx = (chk * 256 + tid) * 16;
    int row = idx >> 9, c0 = idx & 511;
    const __bf16* src = ao + (size_t)(m0 + row) * CH + c0;
    __bf16* dst = &Afull[row][c0];
#if HAVE_ASYNC_LDS
    async_copy_b128(src, dst);
    async_copy_b128(src + 8, dst + 8);
#else
#pragma unroll
    for (int i = 0; i < 16; ++i) dst[i] = src[i];
#endif
  }
#pragma unroll
  for (int q = 0; q < 2; ++q) {
    int cch = q * 256 + tid;
    g_s[cch]  = freq_w[cch] + 0.1f * fconv_w[cch];
    bb_s[cch] = 0.1f * fconv_b[cch];
  }
  if (tid < 64) T_s[tid] = Ttab[(m0 + tid) & 1023];
#if HAVE_ASYNC_LDS
  wait_async0();
#endif
  __syncthreads();

  // ---- Fourier-branch LayerNorm stats
  {
    int tok = tid >> 2, part = tid & 3;
    float Tv = T_s[tok];
    const float* xr = x + (size_t)(m0 + tok) * CH + part * 128;
    float s = 0.f, s2 = 0.f;
    for (int i = 0; i < 128; ++i) {
      int cch = part * 128 + i;
      float v = g_s[cch] * xr[i] + bb_s[cch] * Tv;
      s += v; s2 += v * v;
    }
    red[tid] = s;
    __syncthreads();
    if (part == 0) mu_s[tok] = (red[tid] + red[tid + 1] + red[tid + 2] + red[tid + 3]) * (1.f / 512.f);
    __syncthreads();
    red[tid] = s2;
    __syncthreads();
    if (part == 0) {
      float mu = mu_s[tok];
      float var = (red[tid] + red[tid + 1] + red[tid + 2] + red[tid + 3]) * (1.f / 512.f) - mu * mu;
      rs_s[tok] = rsqrtf(var + 1e-5f);
    }
    __syncthreads();
  }

  float w0, w1;
  { float e0 = __expf(mix_w[0]), e1 = __expf(mix_w[1]); float inv = 1.f / (e0 + e1); w0 = e0 * inv; w1 = e1 * inv; }

  int r16 = (wave & 3) * 16;
  int cn  = (wave >> 2) * 32;
  int nloc = lane & 15, half = lane >> 4;

  for (int nb = 0; nb < 8; ++nb) {
    int n0 = nb * 64;
    v8f acc0 = {}, acc1 = {};
    for (int kk = 0; kk < CH; kk += 64) {
      { int n = tid >> 2, c0 = (tid & 3) * 16;
        const float* src = proj_w + (size_t)(n0 + n) * CH + kk + c0;
#pragma unroll
        for (int i = 0; i < 16; ++i) Bt[c0 + i][n] = f2bf(src[i]); }
      __syncthreads();
      v16bf a0 = load_a_frag(&Afull[0][0] + kk,      r16, 512, lane);
      v16bf a1 = load_a_frag(&Afull[0][0] + kk + 32, r16, 512, lane);
      acc0 = wmma_bf16(a0, load_b_frag(&Bt[0][0], 0,  64, cn,      lane), acc0);
      acc0 = wmma_bf16(a1, load_b_frag(&Bt[0][0], 32, 64, cn,      lane), acc0);
      acc1 = wmma_bf16(a0, load_b_frag(&Bt[0][0], 0,  64, cn + 16, lane), acc1);
      acc1 = wmma_bf16(a1, load_b_frag(&Bt[0][0], 32, 64, cn + 16, lane), acc1);
      __syncthreads();
    }
    // fused epilogue: mix(spatial, fourier), hoisted bases
#pragma unroll
    for (int which = 0; which < 2; ++which) {
      int j = n0 + cn + which * 16 + nloc;
      float pb = proj_b[j], lg = ln_g[j], lb = ln_b[j];
      float g = g_s[j], bb = bb_s[j];
      v8f acc = which ? acc1 : acc0;
      const float* xbase = x + (size_t)(m0 + r16 + 8 * half) * CH + j;
      float* obase = out + (size_t)(m0 + r16 + 8 * half) * CH + j;
#pragma unroll
      for (int r = 0; r < 8; ++r) {
        int trow = r16 + r + 8 * half;
        float xf   = g * xbase[(size_t)r * CH] + bb * T_s[trow];
        float four = (xf - mu_s[trow]) * rs_s[trow] * lg + lb;
        float spa  = acc[r] + pb;
        obase[(size_t)r * CH] = w0 * spa + w1 * four;
      }
    }
  }
}

// ---------------------------------------------------------------------------
extern "C" void kernel_launch(void* const* d_in, const int* in_sizes, int n_in,
                              void* d_out, int out_size, void* d_ws, size_t ws_size,
                              hipStream_t stream) {
  const float* x       = (const float*)d_in[0];
  const float* qkv_w   = (const float*)d_in[1];
  const float* proj_w  = (const float*)d_in[2];
  const float* proj_b  = (const float*)d_in[3];
  const float* fconv_w = (const float*)d_in[4];
  const float* fconv_b = (const float*)d_in[5];
  const float* freq_w  = (const float*)d_in[6];
  const float* ln_g    = (const float*)d_in[7];
  const float* ln_b    = (const float*)d_in[8];
  const float* mix_w   = (const float*)d_in[9];
  float* out = (float*)d_out;

  char* ws = (char*)d_ws;
  float*  Ttab = (float*)ws;                                 // 4 KB
  __bf16* qb = (__bf16*)(ws + 4096);                         // 8 MB
  __bf16* kb = qb + (size_t)TTOT * HEADD;                    // 8 MB
  __bf16* vb = kb + (size_t)TTOT * HEADD;                    // 8 MB
  __bf16* ao = vb + (size_t)TTOT * HEADD;                    // 8 MB

  ttab_kernel<<<1, 1024, 0, stream>>>(Ttab);
  qkv_kernel<<<dim3(TTOT / 64, QKV_N / 64), 256, 0, stream>>>(x, qkv_w, qb, kb, vb);
  attn_kernel<<<BATCH * NHEADS * 4, 512, 0, stream>>>(qb, kb, vb, ao);
  final_kernel<<<TTOT / 64, 256, 0, stream>>>(ao, x, proj_w, proj_b, fconv_w, fconv_b,
                                              freq_w, ln_g, ln_b, mix_w, Ttab, out);
}
